// DenseGNN_5_33809982554706
// MI455X (gfx1250) — matrix operand order repaired and checked by reading, not compiled, not run on hardware
//
#include <hip/hip_runtime.h>

#define Bsz 8
#define Nn  1024
#define Jm  3
#define GS  1088   // channel stride of gmul-output buffer (>= 3*352)
#define HS  384    // channel stride of h buffers (>= 352)

typedef __bf16 v16bf __attribute__((ext_vector_type(16)));
typedef __bf16 v8bf  __attribute__((ext_vector_type(8)));
typedef float  v8f   __attribute__((ext_vector_type(8)));
typedef float  f32x4 __attribute__((ext_vector_type(4)));

#if defined(__has_builtin)
#if __has_builtin(__builtin_amdgcn_global_load_async_to_lds_b128)
#define HAVE_ASYNC_LDS 1
#endif
#if __has_builtin(__builtin_amdgcn_s_wait_asynccnt)
#define HAVE_WAIT_ASYNC 1
#endif
#endif

__device__ __forceinline__ unsigned short f2bf(float f) {
  unsigned u = __float_as_uint(f);
  return (unsigned short)((u + 0x7FFFu + ((u >> 16) & 1u)) >> 16);  // RNE
}

#ifdef HAVE_ASYNC_LDS
typedef int v4i_vs __attribute__((vector_size(16)));
// offset/cpol must be constant expressions at Sema time -> template params.
template <int OFF>
__device__ __forceinline__ void async_copy16(const void* g, void* l) {
  auto gp = (__attribute__((address_space(1))) v4i_vs*)(unsigned long long)(size_t)g;
  auto lp = (__attribute__((address_space(3))) v4i_vs*)(unsigned int)(size_t)l;
  __builtin_amdgcn_global_load_async_to_lds_b128(gp, lp, OFF, 0);
}
#endif

__device__ __forceinline__ void wait_async_lds() {
#ifdef HAVE_ASYNC_LDS
#ifdef HAVE_WAIT_ASYNC
  __builtin_amdgcn_s_wait_asynccnt(0);
#else
  asm volatile("s_wait_asynccnt 0" ::: "memory");
#endif
#endif
}

// ---------------------------------------------------------------- utilities
__global__ void cvt_bf16_kernel(const float* __restrict__ s,
                                unsigned short* __restrict__ d, long long n) {
  long long i = (long long)blockIdx.x * blockDim.x + threadIdx.x;
  if (i < n) d[i] = f2bf(s[i]);
}

__global__ void zero_kernel(float* p, int n) {
  int i = blockIdx.x * blockDim.x + threadIdx.x;
  if (i < n) p[i] = 0.f;
}

// X: [B][1024][ldx], channels contiguous. Per-channel sum/sumsq over (b, row).
__global__ void bn_stats_kernel(const float* __restrict__ X, int ldx, int CH,
                                float* __restrict__ sum, float* __restrict__ sumsq) {
  int b = blockIdx.y;
  int r0 = blockIdx.x * 32;
  const float* base = X + ((size_t)b * Nn + r0) * ldx;
  for (int ch = threadIdx.x; ch < CH; ch += blockDim.x) {
    float s = 0.f, q = 0.f;
#pragma unroll 4
    for (int i = 0; i < 32; ++i) {
      float v = base[(size_t)i * ldx + ch];
      s += v; q += v * v;
    }
    atomicAdd(&sum[ch], s);
    atomicAdd(&sumsq[ch], q);
  }
}

__global__ void bn_finalize_kernel(const float* __restrict__ sum,
                                   const float* __restrict__ sumsq,
                                   const float* __restrict__ g,
                                   const float* __restrict__ b,
                                   float* __restrict__ scale,
                                   float* __restrict__ shift,
                                   int CH, float inv_count) {
  int ch = blockIdx.x * blockDim.x + threadIdx.x;
  if (ch >= CH) return;
  float mu = sum[ch] * inv_count;
  float var = sumsq[ch] * inv_count - mu * mu;
  float rs = rsqrtf(var + 1e-5f);
  float sc = g[ch] * rs;
  scale[ch] = sc;
  shift[ch] = b[ch] - mu * sc;
}

// First layer: t[b][j][m] = sum_n W[b,j,m,n] * x[b,n]  (GEMV, memory bound)
__global__ void first_gmul_kernel(const float* __restrict__ W,
                                  const float* __restrict__ x,
                                  float* __restrict__ t) {
  int id = blockIdx.x;            // [0, B*J*1024)
  int m = id & (Nn - 1);
  int bj = id >> 10;
  int b = bj / Jm;
  const float* wrow = W + (size_t)bj * Nn * Nn + (size_t)m * Nn;
  const float* xv = x + (size_t)b * Nn;
  float s = 0.f;
  for (int n = threadIdx.x; n < Nn; n += 256) s += wrow[n] * xv[n];
  __shared__ float red[256];
  red[threadIdx.x] = s;
  __syncthreads();
  for (int st = 128; st > 0; st >>= 1) {
    if (threadIdx.x < st) red[threadIdx.x] += red[threadIdx.x + st];
    __syncthreads();
  }
  if (threadIdx.x == 0) t[(size_t)bj * Nn + m] = red[0];
}

__global__ void first_conv_kernel(const float* __restrict__ t,
                                  const float* __restrict__ fw,
                                  const float* __restrict__ fb,
                                  float* __restrict__ h) {
  int idx = blockIdx.x * blockDim.x + threadIdx.x;
  if (idx >= Bsz * Nn) return;
  int b = idx >> 10, m = idx & (Nn - 1);
  float t0 = t[((size_t)b * Jm + 0) * Nn + m];
  float t1 = t[((size_t)b * Jm + 1) * Nn + m];
  float t2 = t[((size_t)b * Jm + 2) * Nn + m];
  float* hb = h + (size_t)b * HS * Nn;
#pragma unroll 4
  for (int o = 0; o < 64; ++o)
    hb[(size_t)o * Nn + m] = fw[o * 3] * t0 + fw[o * 3 + 1] * t1 + fw[o * 3 + 2] * t2 + fb[o];
}

// Last layer: out[b][n][cls] = sum_k lw[cls,k]*gout[b][n][k] + lb[cls], K=528, no BN.
__global__ void last_conv_kernel(const float* __restrict__ gout,
                                 const float* __restrict__ lw,
                                 const float* __restrict__ lb,
                                 float* __restrict__ out, int K) {
  int idx = blockIdx.x * blockDim.x + threadIdx.x;
  if (idx >= Bsz * Nn) return;
  int b = idx >> 10, n = idx & (Nn - 1);
  const float* row = gout + ((size_t)b * Nn + n) * GS;
  float a0 = lb[0], a1 = lb[1];
  for (int k = 0; k < K; ++k) {
    float v = row[k];
    a0 += lw[k] * v;
    a1 += lw[K + k] * v;
  }
  out[(size_t)idx * 2 + 0] = a0;
  out[(size_t)idx * 2 + 1] = a1;
}

// ------------------------------------------------------------- WMMA NT GEMM
// D[row,col] = sum_k A_bf16[row*lda+k] * fold(B_f32[col*ldb+k]) (+ bias[row])
// fold = relu(scale[k]*x + shift[k]) if BN else x.
// gmul:  z = b*J+j, A advances by aStrideZ per z, D column block offset jr*NC.
// Block tile 128(M) x 128(cols), K step 32; 8 waves in 2(M) x 4(col) grid,
// wave tile 64 x 32 -> 4 M-frags x 2 C-frags = 8 WMMA per K-step.
#define BM 128
#define BC 128
#define KC 32
#define SA 40   // padded A row stride (halves)
#define SB 48   // padded B row stride (halves)

template <bool BN>
__global__ __launch_bounds__(256) void gemm_nt_kernel(
    const unsigned short* __restrict__ A, long long aStrideZ, int lda,
    const float* __restrict__ Bsrc, long long bStrideB, int ldb,
    float* __restrict__ D, long long dStrideB, int drs, int dcs,
    const float* __restrict__ bias,
    const float* __restrict__ scale, const float* __restrict__ shift,
    int M, int NC, int K, int jdiv) {
  __shared__ __attribute__((aligned(32))) unsigned short sA[BM * SA];
  __shared__ __attribute__((aligned(32))) unsigned short sB[BC * SB];

  const int tid = threadIdx.x;
  const int lane = tid & 31, wave = tid >> 5;
  const int mg = wave & 1, cg = wave >> 1;          // wave grid: 2 (M) x 4 (cols)
  const int half = lane >> 4, l16 = lane & 15;

  const int z = blockIdx.z;
  const int b = z / jdiv, jr = z % jdiv;
  const int m0 = blockIdx.y * BM, c0 = blockIdx.x * BC;

  const unsigned short* Ab = A + (size_t)z * (size_t)aStrideZ;
  const float* Bb = Bsrc + (size_t)b * (size_t)bStrideB;
  float* Db = D + (size_t)b * (size_t)dStrideB + (size_t)jr * (size_t)NC * (size_t)dcs;

  v8f acc[4][2];
  const v8f vzero = {0.f, 0.f, 0.f, 0.f, 0.f, 0.f, 0.f, 0.f};
#pragma unroll
  for (int i = 0; i < 4; ++i) { acc[i][0] = vzero; acc[i][1] = vzero; }

  const int KT = K / KC;                  // K is a multiple of 32 for every call
  const int arow = tid >> 1, ahalf = tid & 1;   // A: 128 rows x 2 x 16 halves
  const int brow = tid >> 1, bhalf = tid & 1;   // B: 128 rows x 2 x 16 floats

  for (int kt = 0; kt < KT; ++kt) {
    const int k0 = kt * KC;
    __syncthreads();

    // ---- stage A (already bf16): 32 bytes per thread
    {
      const int grow = m0 + arow;
#ifdef HAVE_ASYNC_LDS
      if (grow < M) {
        const unsigned short* g = Ab + (size_t)grow * lda + k0 + ahalf * 16;
        unsigned short* l = &sA[arow * SA + ahalf * 16];
        async_copy16<0>(g, l);
        async_copy16<16>(g, l);
        if (kt + 1 < KT) __builtin_prefetch(g + KC, 0, 3);
      }
      // rows >= M keep stale LDS; their D rows are never stored.
#else
      uint4 v0 = make_uint4(0, 0, 0, 0), v1 = make_uint4(0, 0, 0, 0);
      if (grow < M) {
        const uint4* s = (const uint4*)(Ab + (size_t)grow * lda + k0 + ahalf * 16);
        v0 = s[0]; v1 = s[1];
        if (kt + 1 < KT)
          __builtin_prefetch(Ab + (size_t)grow * lda + k0 + KC + ahalf * 16, 0, 3);
      }
      uint4* d = (uint4*)&sA[arow * SA + ahalf * 16];
      d[0] = v0; d[1] = v1;
#endif
    }
    // ---- stage B (fp32 -> bf16, optional per-k BN+ReLU): 16 floats per thread
    {
      const int gcol = c0 + brow;
      f32x4 w0 = {0.f, 0.f, 0.f, 0.f}, w1 = w0, w2 = w0, w3 = w0;
      if (gcol < NC) {
        const float* bs = Bb + (size_t)gcol * ldb + k0 + bhalf * 16;
        w0 = *(const f32x4*)bs;
        w1 = *(const f32x4*)(bs + 4);
        w2 = *(const f32x4*)(bs + 8);
        w3 = *(const f32x4*)(bs + 12);
        if (kt + 1 < KT)
          __builtin_prefetch(Bb + (size_t)gcol * ldb + k0 + KC + bhalf * 16, 0, 3);
      }
      unsigned short tmp[16];
#pragma unroll
      for (int q = 0; q < 16; ++q) {
        float xv = (q < 4) ? w0[q] : (q < 8) ? w1[q - 4] : (q < 12) ? w2[q - 8] : w3[q - 12];
        if (BN) {
          const int kk = k0 + bhalf * 16 + q;
          xv = fmaxf(xv * scale[kk] + shift[kk], 0.f);
        }
        tmp[q] = f2bf(xv);
      }
      uint4 p0, p1;
      p0.x = (unsigned)tmp[0] | ((unsigned)tmp[1] << 16);
      p0.y = (unsigned)tmp[2] | ((unsigned)tmp[3] << 16);
      p0.z = (unsigned)tmp[4] | ((unsigned)tmp[5] << 16);
      p0.w = (unsigned)tmp[6] | ((unsigned)tmp[7] << 16);
      p1.x = (unsigned)tmp[8] | ((unsigned)tmp[9] << 16);
      p1.y = (unsigned)tmp[10] | ((unsigned)tmp[11] << 16);
      p1.z = (unsigned)tmp[12] | ((unsigned)tmp[13] << 16);
      p1.w = (unsigned)tmp[14] | ((unsigned)tmp[15] << 16);
      unsigned short* dst = &sB[brow * SB + bhalf * 16];
      *(uint4*)dst = p0;
      *(uint4*)(dst + 8) = p1;
    }
    wait_async_lds();
    __syncthreads();

    // ---- compute: 2 B frags per wave, 4 A frags reused across both
    const v16bf bf0 = __builtin_shufflevector(
        *(const v8bf*)&sB[(cg * 32 + l16) * SB + half * 8],
        *(const v8bf*)&sB[(cg * 32 + l16) * SB + 16 + half * 8],
        0, 1, 2, 3, 4, 5, 6, 7, 8, 9, 10, 11, 12, 13, 14, 15);
    const v16bf bf1 = __builtin_shufflevector(
        *(const v8bf*)&sB[(cg * 32 + 16 + l16) * SB + half * 8],
        *(const v8bf*)&sB[(cg * 32 + 16 + l16) * SB + 16 + half * 8],
        0, 1, 2, 3, 4, 5, 6, 7, 8, 9, 10, 11, 12, 13, 14, 15);
#pragma unroll
    for (int mf = 0; mf < 4; ++mf) {
      const int r = mg * 64 + mf * 16 + l16;
      const v8bf lo = *(const v8bf*)&sA[r * SA + half * 8];
      const v8bf hi = *(const v8bf*)&sA[r * SA + 16 + half * 8];
      const v16bf afrag = __builtin_shufflevector(lo, hi, 0, 1, 2, 3, 4, 5, 6, 7,
                                                  8, 9, 10, 11, 12, 13, 14, 15);
      acc[mf][0] = __builtin_amdgcn_wmma_f32_16x16x32_bf16(
          false, afrag, false, bf0, (short)0, acc[mf][0], false, false);
      acc[mf][1] = __builtin_amdgcn_wmma_f32_16x16x32_bf16(
          false, afrag, false, bf1, (short)0, acc[mf][1], false, false);
    }
  }

  // ---- store D fragments (row = g + 8*half within frag, col = l16)
#pragma unroll
  for (int cf = 0; cf < 2; ++cf) {
    const int col = c0 + cg * 32 + cf * 16 + l16;
    if (col >= NC) continue;
#pragma unroll
    for (int mf = 0; mf < 4; ++mf) {
      union { v8f v; float f[8]; } u;
      u.v = acc[mf][cf];
#pragma unroll
      for (int g = 0; g < 8; ++g) {
        const int row = m0 + mg * 64 + mf * 16 + g + 8 * half;
        if (row < M) {
          float v = u.f[g];
          if (bias) v += bias[row];
          Db[(size_t)row * drs + (size_t)col * dcs] = v;
        }
      }
    }
  }
}

// ---------------------------------------------------------------- host side
extern "C" void kernel_launch(void* const* d_in, const int* in_sizes, int n_in,
                              void* d_out, int out_size, void* d_ws, size_t ws_size,
                              hipStream_t stream) {
  (void)in_sizes; (void)n_in; (void)out_size; (void)ws_size;
  const float* W = (const float*)d_in[0];
  const float* x = (const float*)d_in[1];
  int pi = 2;
  auto P = [&](int i) { return (const float*)d_in[i]; };

  struct BlockP { const float *bn1g, *bn1b, *c1w, *c1b, *bn2g, *bn2b, *c2w, *c2b; };
  struct TransP { const float *bng, *bnb, *cw, *cb; };
  const float* first_w = P(pi++); const float* first_b = P(pi++);
  BlockP blk[9]; TransP tr[3];
  int bi = 0;
  for (int s = 0; s < 3; ++s) {
    for (int i = 0; i < 3; ++i) {
      blk[bi] = {P(pi), P(pi + 1), P(pi + 2), P(pi + 3),
                 P(pi + 4), P(pi + 5), P(pi + 6), P(pi + 7)};
      pi += 8; ++bi;
    }
    tr[s] = {P(pi), P(pi + 1), P(pi + 2), P(pi + 3)};
    pi += 4;
  }
  const float* last_w = P(pi++); const float* last_b = P(pi++);

  // workspace carve
  char* wp = (char*)d_ws;
  auto alloc = [&](size_t bytes) {
    void* p = (void*)wp;
    wp += (bytes + 255) & ~(size_t)255;
    return p;
  };
  unsigned short* Wbf = (unsigned short*)alloc((size_t)Bsz * Jm * Nn * Nn * 2);  // 48 MB
  float* gout = (float*)alloc((size_t)Bsz * Nn * GS * 4);                        // 35.7 MB
  float* h0 = (float*)alloc((size_t)Bsz * HS * Nn * 4);                          // 12.6 MB
  float* h1 = (float*)alloc((size_t)Bsz * HS * Nn * 4);                          // 12.6 MB
  float* obuf = (float*)alloc((size_t)Bsz * Nn * 256 * 4);                       // 8.4 MB
  unsigned short* wcv = (unsigned short*)alloc((size_t)1 << 20);                 // 1 MB
  float* stats = (float*)alloc((size_t)4 * GS * 4);
  float* ssum = stats, *ssq = stats + GS, *sc = stats + 2 * GS, *sh = stats + 3 * GS;
  float* tbuf = (float*)alloc((size_t)Bsz * Jm * Nn * 4);

  auto cvt = [&](const float* src, unsigned short* dst, long long n) {
    cvt_bf16_kernel<<<(unsigned)((n + 255) / 256), 256, 0, stream>>>(src, dst, n);
  };
  auto bnstats = [&](const float* X, int ldx, int CH, const float* g, const float* bb) {
    zero_kernel<<<(2 * GS + 255) / 256, 256, 0, stream>>>(ssum, 2 * GS);
    bn_stats_kernel<<<dim3(Nn / 32, Bsz), 256, 0, stream>>>(X, ldx, CH, ssum, ssq);
    bn_finalize_kernel<<<(CH + 255) / 256, 256, 0, stream>>>(
        ssum, ssq, g, bb, sc, sh, CH, 1.0f / (Bsz * Nn));
  };
  auto gemm = [&](bool bn, const unsigned short* A, long long aZ, int lda,
                  const float* Bp, long long bB, int ldb, float* Dp, long long dB,
                  int drs, int dcs, const float* bias, int M, int NC, int K,
                  int jdiv, int zcnt) {
    dim3 grid((NC + BC - 1) / BC, (M + BM - 1) / BM, zcnt);
    if (bn)
      gemm_nt_kernel<true><<<grid, 256, 0, stream>>>(
          A, aZ, lda, Bp, bB, ldb, Dp, dB, drs, dcs, bias, sc, sh, M, NC, K, jdiv);
    else
      gemm_nt_kernel<false><<<grid, 256, 0, stream>>>(
          A, aZ, lda, Bp, bB, ldb, Dp, dB, drs, dcs, bias, sc, sh, M, NC, K, jdiv);
  };

  const long long hBS = (long long)HS * Nn;       // h per-batch stride
  const long long gBS = (long long)Nn * GS;       // gout per-batch stride
  const long long oBS = (long long)Nn * 256;      // obuf per-batch stride
  const long long wZS = (long long)Nn * Nn;       // Wbf per-(b,j) stride

  // W -> bf16 once (L2-resident thereafter: 50 MB << 192 MB L2)
  cvt(W, Wbf, (long long)Bsz * Jm * Nn * Nn);

  // first layer
  first_gmul_kernel<<<Bsz * Jm * Nn, 256, 0, stream>>>(W, x, tbuf);
  first_conv_kernel<<<(Bsz * Nn + 255) / 256, 256, 0, stream>>>(tbuf, first_w, first_b, h0);

  float* hc = h0; float* hn = h1;
  int c = 64; bi = 0;
  for (int s = 0; s < 3; ++s) {
    for (int i = 0; i < 3; ++i) {
      const BlockP& p = blk[bi++];
      const int JC = 3 * c;
      // gmul: gout[b][m][j*c + cc]
      gemm(false, Wbf, wZS, Nn, hc, hBS, Nn, gout, gBS, GS, 1, nullptr,
           Nn, c, Nn, Jm, Bsz * Jm);
      bnstats(gout, GS, JC, p.bn1g, p.bn1b);
      // conv1: [256,JC] x bn(gout) -> obuf[b][n][256]
      cvt(p.c1w, wcv, (long long)256 * JC);
      gemm(true, wcv, 0, JC, gout, gBS, GS, obuf, oBS, 1, 256, p.c1b,
           256, Nn, JC, 1, Bsz);
      bnstats(obuf, 256, 256, p.bn2g, p.bn2b);
      // conv2: [64,256] x bn(obuf) -> append into hc at channel offset c
      cvt(p.c2w, wcv, 64 * 256);
      gemm(true, wcv, 0, 256, obuf, oBS, 256, hc + (size_t)c * Nn, hBS, Nn, 1,
           p.c2b, 64, Nn, 256, 1, Bsz);
      c += 64;
    }
    const TransP& t = tr[s];
    const int JC = 3 * c, co = c / 2;
    gemm(false, Wbf, wZS, Nn, hc, hBS, Nn, gout, gBS, GS, 1, nullptr,
         Nn, c, Nn, Jm, Bsz * Jm);
    bnstats(gout, GS, JC, t.bng, t.bnb);
    cvt(t.cw, wcv, (long long)co * JC);
    gemm(true, wcv, 0, JC, gout, gBS, GS, hn, hBS, Nn, 1, t.cb,
         co, Nn, JC, 1, Bsz);
    float* tmp = hc; hc = hn; hn = tmp;
    c = co;
  }

  // last layer: gmul (c=176) then 2-class conv, output [B,N,2]
  gemm(false, Wbf, wZS, Nn, hc, hBS, Nn, gout, gBS, GS, 1, nullptr,
       Nn, c, Nn, Jm, Bsz * Jm);
  last_conv_kernel<<<(Bsz * Nn + 255) / 256, 256, 0, stream>>>(
      gout, last_w, last_b, (float*)d_out, Jm * c);
}
